// SetAbstraction_2989297238646
// MI455X (gfx1250) — matrix operand order repaired and checked by reading, not compiled, not run on hardware
//
#include <hip/hip_runtime.h>

#define BB 2
#define NPER 16384
#define FDIM 32
#define RRAD 0.2f
#define R2 0.04f
#define KN 64
#define H1 64
#define COUT 128
#define MM 4096
#define WAVES 4

typedef _Float16 v16h __attribute__((ext_vector_type(16)));
typedef float    v8f  __attribute__((ext_vector_type(8)));

// ---------------- Kernel 1: pre-swizzle W1/W2 into WMMA B-fragment layout ----------------
// Fragment storage: block = n*2 + k (k = 32-wide K step, n = 16-wide N tile);
// offset = (block*32 + lane)*16 + j ; value = W[k*32 + (lane>>4)*16 + j][n*16 + (lane&15)]
__global__ __launch_bounds__(256) void prep_weights_kernel(
    const float* __restrict__ W1, const float* __restrict__ W2,
    _Float16* __restrict__ w1f, _Float16* __restrict__ w2f) {
  int e = blockIdx.x * 256 + threadIdx.x;
  if (e < 4096) {                       // W1: (35->64 padded) x 64, 8 blocks
    int j = e & 15, lane = (e >> 4) & 31, blk = e >> 9;
    int k = blk & 1, n = blk >> 1;      // n in 0..3
    int kk = k * 32 + ((lane >> 4) << 4) + j;
    int col = (n << 4) + (lane & 15);
    float v = (kk < FDIM + 3) ? W1[kk * H1 + col] : 0.0f;
    w1f[e] = (_Float16)v;
  } else if (e < 4096 + 8192) {         // W2: 64 x 128, 16 blocks
    int e2 = e - 4096;
    int j = e2 & 15, lane = (e2 >> 4) & 31, blk = e2 >> 9;
    int k = blk & 1, n = blk >> 1;      // n in 0..7
    int kk = k * 32 + ((lane >> 4) << 4) + j;
    int col = (n << 4) + (lane & 15);
    w2f[e2] = (_Float16)W2[kk * COUT + col];
  }
}

// ---------------- Kernel 2: farthest point sampling (1 workgroup per cloud) ----------------
__global__ __launch_bounds__(1024) void fps_kernel(
    const float* __restrict__ pos, int* __restrict__ idxList,
    float* __restrict__ pcOut, int* __restrict__ batchOut) {
  const int b = blockIdx.x;
  const int tid = threadIdx.x;
  const int lane = tid & 31;
  const int wid = tid >> 5;
  const float* pb = pos + (size_t)b * NPER * 3;

  __shared__ float rv[32];
  __shared__ int   ri[32];
  __shared__ int   swidx;

  // 16 consecutive points per thread, cached in VGPRs
  float px[16], py[16], pz[16], dd[16];
  const int base = tid * 16;
  const float x0 = pb[0], y0 = pb[1], z0 = pb[2];
#pragma unroll
  for (int q = 0; q < 16; ++q) {
    px[q] = pb[(base + q) * 3 + 0];
    py[q] = pb[(base + q) * 3 + 1];
    pz[q] = pb[(base + q) * 3 + 2];
    float dx = px[q] - x0, dy = py[q] - y0, dz = pz[q] - z0;
    dd[q] = dx * dx + dy * dy + dz * dz;
  }
  for (int t = tid; t < MM; t += 1024) batchOut[b * MM + t] = b;
  if (tid == 0) {
    idxList[b * MM] = 0;
    pcOut[(size_t)(b * MM) * 3 + 0] = x0;
    pcOut[(size_t)(b * MM) * 3 + 1] = y0;
    pcOut[(size_t)(b * MM) * 3 + 2] = z0;
  }

  for (int m = 1; m < MM; ++m) {
    // local argmax (first-index tie-break: strict >)
    float bv = dd[0]; int bi = base;
#pragma unroll
    for (int q = 1; q < 16; ++q)
      if (dd[q] > bv) { bv = dd[q]; bi = base + q; }
    // wave reduce
#pragma unroll
    for (int off = 16; off > 0; off >>= 1) {
      float vo = __shfl_down(bv, off, 32);
      int   io = __shfl_down(bi, off, 32);
      if (vo > bv || (vo == bv && io < bi)) { bv = vo; bi = io; }
    }
    if (lane == 0) { rv[wid] = bv; ri[wid] = bi; }
    __syncthreads();
    if (wid == 0) {
      float v = rv[lane]; int i = ri[lane];
#pragma unroll
      for (int off = 16; off > 0; off >>= 1) {
        float vo = __shfl_down(v, off, 32);
        int   io = __shfl_down(i, off, 32);
        if (vo > v || (vo == v && io < i)) { v = vo; i = io; }
      }
      if (lane == 0) swidx = i;
    }
    __syncthreads();
    const int wi = swidx;
    const float cx = pb[wi * 3 + 0], cy = pb[wi * 3 + 1], cz = pb[wi * 3 + 2];
    if (tid == 0) {
      idxList[b * MM + m] = wi;
      pcOut[(size_t)(b * MM + m) * 3 + 0] = cx;
      pcOut[(size_t)(b * MM + m) * 3 + 1] = cy;
      pcOut[(size_t)(b * MM + m) * 3 + 2] = cz;
    }
#pragma unroll
    for (int q = 0; q < 16; ++q) {
      float dx = px[q] - cx, dy = py[q] - cy, dz = pz[q] - cz;
      float d2 = dx * dx + dy * dy + dz * dz;
      dd[q] = fminf(dd[q], d2);
    }
  }
}

// ---------------- WMMA fragment loaders (contiguous k = 16*(lane>>4) + s in both A and B) ----------------
__device__ __forceinline__ v16h ldsA(const _Float16* tile, int r, int kbase, int lane) {
  const int row = (r << 4) + (lane & 15);
  const int h = lane >> 4;
  const _Float16* p = tile + row * 64 + kbase + h * 16;
  v16h a;
#pragma unroll
  for (int j = 0; j < 8; ++j) a[j] = p[j];
  const _Float16* q = p + 8;
#pragma unroll
  for (int j = 0; j < 8; ++j) a[8 + j] = q[j];
  return a;
}
__device__ __forceinline__ v16h ldsB(const _Float16* frags, int k, int n, int lane) {
  const _Float16* p = frags + ((((n << 1) + k) * 32 + lane) << 4);
  v16h b;
#pragma unroll
  for (int j = 0; j < 16; ++j) b[j] = p[j];
  return b;
}

// ---------------- Kernel 3: ball query + gather + fused WMMA MLP + max-pool ----------------
__global__ __launch_bounds__(128) void sa_kernel(
    const float* __restrict__ x, const float* __restrict__ pos,
    const int* __restrict__ idxList,
    const _Float16* __restrict__ w1f, const _Float16* __restrict__ w2f,
    const float* __restrict__ b1, const float* __restrict__ b2,
    float* __restrict__ out) {
  __shared__ _Float16 sW1[4096];
  __shared__ _Float16 sW2[8192];
  __shared__ float sb1[H1];
  __shared__ float sb2[COUT];
  __shared__ int   snbr[WAVES][KN];
  __shared__ _Float16 smsg[WAVES][64 * 64];

  const int tid = threadIdx.x, lane = tid & 31, w = tid >> 5;
  for (int i = tid; i < 4096; i += 128) sW1[i] = w1f[i];
  for (int i = tid; i < 8192; i += 128) sW2[i] = w2f[i];
  if (tid < H1) sb1[tid] = b1[tid];
  if (tid < COUT) sb2[tid] = b2[tid];
  __syncthreads();

  const int c = blockIdx.x * WAVES + w;       // global centroid id in [0, B*M)
  const int b = c / MM;
  const int ci = idxList[c];
  const float* pb = pos + (size_t)b * NPER * 3;
  const float* xb = x + (size_t)b * NPER * FDIM;
  const float cx = pb[ci * 3 + 0], cy = pb[ci * 3 + 1], cz = pb[ci * 3 + 2];

  // ---- ball query: first K neighbors with d2 <= R^2 (wave-uniform control) ----
  int cnt = 0;
  for (int basep = 0; basep < NPER; basep += 32) {
    const int j = basep + lane;
    float dx = pb[j * 3 + 0] - cx, dy = pb[j * 3 + 1] - cy, dz = pb[j * 3 + 2] - cz;
    float d2 = dx * dx + dy * dy + dz * dz;
    bool pred = (d2 <= R2);
    unsigned mask = __builtin_amdgcn_ballot_w32(pred);
    int ofs = cnt + __popc(mask & ((1u << lane) - 1u));
    if (pred && ofs < KN) snbr[w][ofs] = j;
    cnt += __popc(mask);
    if (cnt >= KN) break;
  }
  cnt = (cnt < KN) ? cnt : KN;

  // ---- build msg tile (64 rows x 64 f16 cols; cols 35..63 zero-padded) ----
  // faithful to source: pos part = pj - pc/R
  const float cxr = cx * 5.0f, cyr = cy * 5.0f, czr = cz * 5.0f;
  _Float16* tile = smsg[w];
  for (int k2 = lane; k2 < KN; k2 += 32) {
    _Float16* row = tile + k2 * 64;
    if (k2 < cnt) {
      const int p = snbr[w][k2];
      const float* xr = xb + (size_t)p * FDIM;
#pragma unroll
      for (int f = 0; f < FDIM; ++f) row[f] = (_Float16)xr[f];
      row[32] = (_Float16)(pb[p * 3 + 0] - cxr);
      row[33] = (_Float16)(pb[p * 3 + 1] - cyr);
      row[34] = (_Float16)(pb[p * 3 + 2] - czr);
#pragma unroll
      for (int f = 35; f < 64; ++f) row[f] = (_Float16)0.0f;
    } else {
#pragma unroll
      for (int f = 0; f < 64; ++f) row[f] = (_Float16)0.0f;
    }
  }

  // ---- GEMM1 (64x64 @ 64x64), bias + ReLU, h1 overwrites msg rows per strip ----
#pragma unroll
  for (int r = 0; r < 4; ++r) {
    v16h a0 = ldsA(tile, r, 0, lane);
    v16h a1 = ldsA(tile, r, 32, lane);
    v8f accs[4];
#pragma unroll
    for (int n = 0; n < 4; ++n) {
      v8f acc = {};
      acc = __builtin_amdgcn_wmma_f32_16x16x32_f16(false, a0, false, ldsB(sW1, 0, n, lane),
                                                   (short)0, acc, false, false);
      acc = __builtin_amdgcn_wmma_f32_16x16x32_f16(false, a1, false, ldsB(sW1, 1, n, lane),
                                                   (short)0, acc, false, false);
      accs[n] = acc;
    }
    const int rh = (lane >> 4) << 3;
#pragma unroll
    for (int n = 0; n < 4; ++n) {
      const int col = (n << 4) + (lane & 15);
      const float bv = sb1[col];
#pragma unroll
      for (int j = 0; j < 8; ++j) {
        const int row = (r << 4) + rh + j;
        float v = accs[n][j] + bv;
        tile[row * 64 + col] = (_Float16)fmaxf(v, 0.0f);
      }
    }
  }

  // ---- GEMM2 (64x64 @ 64x128) fused with masked column-max over valid rows ----
  float mcol[8];
#pragma unroll
  for (int n = 0; n < 8; ++n) mcol[n] = -3.0e38f;
#pragma unroll
  for (int r = 0; r < 4; ++r) {
    v16h a0 = ldsA(tile, r, 0, lane);
    v16h a1 = ldsA(tile, r, 32, lane);
    const int rb = (r << 4) + ((lane >> 4) << 3);
#pragma unroll
    for (int n = 0; n < 8; ++n) {
      v8f acc = {};
      acc = __builtin_amdgcn_wmma_f32_16x16x32_f16(false, a0, false, ldsB(sW2, 0, n, lane),
                                                   (short)0, acc, false, false);
      acc = __builtin_amdgcn_wmma_f32_16x16x32_f16(false, a1, false, ldsB(sW2, 1, n, lane),
                                                   (short)0, acc, false, false);
#pragma unroll
      for (int j = 0; j < 8; ++j)
        if (rb + j < cnt) mcol[n] = fmaxf(mcol[n], acc[j]);
    }
  }
#pragma unroll
  for (int n = 0; n < 8; ++n) {
    float o = __shfl_xor(mcol[n], 16, 32);
    float mm = fmaxf(mcol[n], o);
    if (lane < 16) {
      const int col = (n << 4) + lane;
      float res = (mm > -1.0e37f) ? (mm + sb2[col]) : 0.0f;
      out[(size_t)c * COUT + col] = res;
    }
  }
}

// ---------------- host launcher ----------------
extern "C" void kernel_launch(void* const* d_in, const int* in_sizes, int n_in,
                              void* d_out, int out_size, void* d_ws, size_t ws_size,
                              hipStream_t stream) {
  const float* x   = (const float*)d_in[0];
  const float* pos = (const float*)d_in[1];
  // d_in[2] = batch (unused; reconstructed)
  const float* W1 = (const float*)d_in[3];
  const float* b1 = (const float*)d_in[4];
  const float* W2 = (const float*)d_in[5];
  const float* b2 = (const float*)d_in[6];

  float* out   = (float*)d_out;                       // (B*M, COUT)
  float* pcOut = out + (size_t)BB * MM * COUT;        // (B*M, 3)
  int* batchOut = (int*)(pcOut + (size_t)BB * MM * 3);// (B*M,)

  int* idxList = (int*)d_ws;                                   // B*M ints
  _Float16* w1f = (_Float16*)((char*)d_ws + 32768);            // 4096 halves
  _Float16* w2f = w1f + 4096;                                  // 8192 halves

  prep_weights_kernel<<<48, 256, 0, stream>>>(W1, W2, w1f, w2f);
  fps_kernel<<<BB, 1024, 0, stream>>>(pos, idxList, pcOut, batchOut);
  sa_kernel<<<(BB * MM) / WAVES, 128, 0, stream>>>(x, pos, idxList, w1f, w2f, b1, b2, out);
}
